// CausalSelfAttention_26938034880888
// MI455X (gfx1250) — matrix operand order repaired and checked by reading, not compiled, not run on hardware
//
#include <hip/hip_runtime.h>

#define Bsz 4
#define Tn  2048
#define Dm  768
#define NHh 12
#define DHd 64

typedef __attribute__((ext_vector_type(16))) __bf16 v16bf;
typedef __attribute__((ext_vector_type(8)))  __bf16 v8bf;
typedef __attribute__((ext_vector_type(8)))  float  v8f;

union V16 { v16bf v; v8bf h[2]; };

__device__ __forceinline__ __bf16 f2bf(float f) {
  unsigned u = __builtin_bit_cast(unsigned, f);
  unsigned r = u + 0x7FFFu + ((u >> 16) & 1u);   // round-to-nearest-even
  unsigned short s = (unsigned short)(r >> 16);
  return __builtin_bit_cast(__bf16, s);
}

__device__ __forceinline__ v8f wmma_bf16(v16bf a, v16bf b, v8f c) {
  // (neg_a, A, neg_b, B, c_mod, C, reuse_a, reuse_b)
  return __builtin_amdgcn_wmma_f32_16x16x32_bf16(false, a, false, b, (short)0, c, false, false);
}

__global__ void cvt_f32_bf16(const float* __restrict__ in, __bf16* __restrict__ out, int n) {
  int i = blockIdx.x * blockDim.x + threadIdx.x;
  if (i < n) out[i] = f2bf(in[i]);
}

// ---------------------------------------------------------------------------
// QKV projection: y = x @ W^T + b  (M=B*T=8192, N=768, K=768)
// blockIdx.z selects Q/K/V. Q,K stored [B,NH,T,DH]; V stored transposed [B,NH,DH,T].
// 4 waves per block; each wave owns a 32x64 output tile.
// ---------------------------------------------------------------------------
__global__ __launch_bounds__(128) void qkv_gemm_wmma(
    const __bf16* __restrict__ xb,
    const __bf16* __restrict__ wqb, const __bf16* __restrict__ wkb, const __bf16* __restrict__ wvb,
    const float* __restrict__ bq, const float* __restrict__ bk, const float* __restrict__ bv,
    __bf16* __restrict__ qo, __bf16* __restrict__ ko, __bf16* __restrict__ vto)
{
  const int lane = threadIdx.x & 31;
  const int wid  = threadIdx.x >> 5;
  const int l15  = lane & 15;
  const int g    = lane >> 4;
  const int z    = blockIdx.z;
  const int mbase = blockIdx.x * 128 + wid * 32;
  const int nbase = blockIdx.y * 64;

  const __bf16* W    = (z == 0) ? wqb : (z == 1) ? wkb : wvb;
  const float*  bias = (z == 0) ? bq  : (z == 1) ? bk  : bv;

  v8f acc[2][4] = {};
  for (int kb = 0; kb < Dm; kb += 32) {
    V16 a[2], bm[4];
#pragma unroll
    for (int mt = 0; mt < 2; ++mt) {
      const __bf16* p = xb + (size_t)(mbase + mt * 16 + l15) * Dm + kb + g * 16;
      a[mt].h[0] = *(const v8bf*)(p);
      a[mt].h[1] = *(const v8bf*)(p + 8);
    }
#pragma unroll
    for (int nt = 0; nt < 4; ++nt) {
      const __bf16* p = W + (size_t)(nbase + nt * 16 + l15) * Dm + kb + g * 16;
      bm[nt].h[0] = *(const v8bf*)(p);
      bm[nt].h[1] = *(const v8bf*)(p + 8);
    }
#pragma unroll
    for (int mt = 0; mt < 2; ++mt)
#pragma unroll
      for (int nt = 0; nt < 4; ++nt)
        acc[mt][nt] = wmma_bf16(a[mt].v, bm[nt].v, acc[mt][nt]);
  }

#pragma unroll
  for (int nt = 0; nt < 4; ++nt) {
    const int col = nbase + nt * 16 + l15;
    const float bval = bias[col];
    const int hh = col >> 6;     // head
    const int dd = col & 63;     // dim in head
#pragma unroll
    for (int mt = 0; mt < 2; ++mt) {
#pragma unroll
      for (int j = 0; j < 8; ++j) {
        const int row = mbase + mt * 16 + g * 8 + j;  // global row = b*T + t
        const int bi = row >> 11;
        const int tt = row & 2047;
        const float v = acc[mt][nt][j] + bval;
        if (z == 2)
          vto[((size_t)(bi * NHh + hh) * DHd + dd) * Tn + tt] = f2bf(v);
        else if (z == 0)
          qo[((size_t)(bi * NHh + hh) * Tn + tt) * DHd + dd] = f2bf(v);
        else
          ko[((size_t)(bi * NHh + hh) * Tn + tt) * DHd + dd] = f2bf(v);
      }
    }
  }
}

// ---------------------------------------------------------------------------
// Flash attention: 4 waves/block, 16 query rows per wave, 32-key chunks.
// Streaming (online) softmax with masked value -100000 after 1/sqrt(64) scale.
// ---------------------------------------------------------------------------
__global__ __launch_bounds__(128) void flash_attn_wmma(
    const __bf16* __restrict__ q,
    const __bf16* __restrict__ kmat,
    const __bf16* __restrict__ vtm,
    __bf16* __restrict__ zb)
{
  __shared__ alignas(16) __bf16 pst[4][16][40];   // padded rows (bank spread)
  const int lane = threadIdx.x & 31;
  const int wid  = threadIdx.x >> 5;
  const int l15  = lane & 15;
  const int g    = lane >> 4;
  const int bh   = blockIdx.y;
  const int bi   = bh / NHh;
  const int hh   = bh % NHh;
  const int qbase = blockIdx.x * 64 + wid * 16;

  const __bf16* Qp = q    + (size_t)bh * Tn * DHd;
  const __bf16* Kp = kmat + (size_t)bh * Tn * DHd;
  const __bf16* Vp = vtm  + (size_t)bh * DHd * Tn;   // [DH, T] transposed

  V16 Qa[2];
#pragma unroll
  for (int d2 = 0; d2 < 2; ++d2) {
    const __bf16* p = Qp + (size_t)(qbase + l15) * DHd + d2 * 32 + g * 16;
    Qa[d2].h[0] = *(const v8bf*)(p);
    Qa[d2].h[1] = *(const v8bf*)(p + 8);
  }

  v8f acc[4] = {};
  float rmax[8], rsum[8];
#pragma unroll
  for (int j = 0; j < 8; ++j) { rmax[j] = -3.0e38f; rsum[j] = 0.0f; }

  const int kend = blockIdx.x * 64 + 64;   // uniform per block; extra chunks fully masked
  for (int k0 = 0; k0 < kend; k0 += 32) {
    // S = Q K^T for 32 keys: 2 key sub-tiles x 2 K-steps over DH
    v8f s[2] = {};
#pragma unroll
    for (int t16 = 0; t16 < 2; ++t16) {
#pragma unroll
      for (int d2 = 0; d2 < 2; ++d2) {
        V16 bk_;
        const __bf16* p = Kp + (size_t)(k0 + t16 * 16 + l15) * DHd + d2 * 32 + g * 16;
        bk_.h[0] = *(const v8bf*)(p);
        bk_.h[1] = *(const v8bf*)(p + 8);
        s[t16] = wmma_bf16(Qa[d2].v, bk_.v, s[t16]);
      }
    }
    // online softmax update (rows live in 16-lane halves; 8 rows per lane)
#pragma unroll
    for (int j = 0; j < 8; ++j) {
      const int row = qbase + g * 8 + j;
      float v0 = s[0][j] * 0.125f;
      float v1 = s[1][j] * 0.125f;
      if (k0 + l15 > row)      v0 = -100000.0f;
      if (k0 + 16 + l15 > row) v1 = -100000.0f;
      float mx = fmaxf(v0, v1);
      mx = fmaxf(mx, __shfl_xor(mx, 1, 16));
      mx = fmaxf(mx, __shfl_xor(mx, 2, 16));
      mx = fmaxf(mx, __shfl_xor(mx, 4, 16));
      mx = fmaxf(mx, __shfl_xor(mx, 8, 16));
      const float nm = fmaxf(rmax[j], mx);
      const float sc = __expf(rmax[j] - nm);
      rmax[j] = nm;
      const float p0 = __expf(v0 - nm);
      const float p1 = __expf(v1 - nm);
      float rs = p0 + p1;
      rs += __shfl_xor(rs, 1, 16);
      rs += __shfl_xor(rs, 2, 16);
      rs += __shfl_xor(rs, 4, 16);
      rs += __shfl_xor(rs, 8, 16);
      rsum[j] = rsum[j] * sc + rs;
#pragma unroll
      for (int dt = 0; dt < 4; ++dt) acc[dt][j] *= sc;
      // stage P (C-layout) into LDS for A-layout reload
      pst[wid][g * 8 + j][l15]      = f2bf(p0);
      pst[wid][g * 8 + j][16 + l15] = f2bf(p1);
    }
    asm volatile("s_wait_dscnt 0x0" ::: "memory");   // intra-wave LDS RAW fence
    V16 Pa;
    Pa.h[0] = *(const v8bf*)(&pst[wid][l15][g * 16]);
    Pa.h[1] = *(const v8bf*)(&pst[wid][l15][g * 16 + 8]);
    // acc += P * V   (B operand from V^T: contiguous 16 keys per lane)
#pragma unroll
    for (int dt = 0; dt < 4; ++dt) {
      V16 bv_;
      const __bf16* p = Vp + (size_t)(dt * 16 + l15) * Tn + k0 + g * 16;
      bv_.h[0] = *(const v8bf*)(p);
      bv_.h[1] = *(const v8bf*)(p + 8);
      acc[dt] = wmma_bf16(Pa.v, bv_.v, acc[dt]);
    }
    asm volatile("s_wait_dscnt 0x0" ::: "memory");   // WAR guard before next chunk's stores
  }

  // normalize and write Z as bf16 [B, T, D]
#pragma unroll
  for (int dt = 0; dt < 4; ++dt) {
#pragma unroll
    for (int j = 0; j < 8; ++j) {
      const int tt = qbase + g * 8 + j;
      const float o = acc[dt][j] / rsum[j];
      zb[((size_t)(bi * Tn + tt)) * Dm + hh * DHd + dt * 16 + l15] = f2bf(o);
    }
  }
}

// ---------------------------------------------------------------------------
// Output projection: out = Z @ Wo^T + bo  (fp32 output)
// ---------------------------------------------------------------------------
__global__ __launch_bounds__(128) void oproj_gemm_wmma(
    const __bf16* __restrict__ zb, const __bf16* __restrict__ wob,
    const float* __restrict__ bo, float* __restrict__ out)
{
  const int lane = threadIdx.x & 31;
  const int wid  = threadIdx.x >> 5;
  const int l15  = lane & 15;
  const int g    = lane >> 4;
  const int mbase = blockIdx.x * 128 + wid * 32;
  const int nbase = blockIdx.y * 64;

  v8f acc[2][4] = {};
  for (int kb = 0; kb < Dm; kb += 32) {
    V16 a[2], bm[4];
#pragma unroll
    for (int mt = 0; mt < 2; ++mt) {
      const __bf16* p = zb + (size_t)(mbase + mt * 16 + l15) * Dm + kb + g * 16;
      a[mt].h[0] = *(const v8bf*)(p);
      a[mt].h[1] = *(const v8bf*)(p + 8);
    }
#pragma unroll
    for (int nt = 0; nt < 4; ++nt) {
      const __bf16* p = wob + (size_t)(nbase + nt * 16 + l15) * Dm + kb + g * 16;
      bm[nt].h[0] = *(const v8bf*)(p);
      bm[nt].h[1] = *(const v8bf*)(p + 8);
    }
#pragma unroll
    for (int mt = 0; mt < 2; ++mt)
#pragma unroll
      for (int nt = 0; nt < 4; ++nt)
        acc[mt][nt] = wmma_bf16(a[mt].v, bm[nt].v, acc[mt][nt]);
  }

#pragma unroll
  for (int nt = 0; nt < 4; ++nt) {
    const int col = nbase + nt * 16 + l15;
    const float bval = bo[col];
#pragma unroll
    for (int mt = 0; mt < 2; ++mt)
#pragma unroll
      for (int j = 0; j < 8; ++j) {
        const int row = mbase + mt * 16 + g * 8 + j;
        out[(size_t)row * Dm + col] = acc[mt][nt][j] + bval;
      }
  }
}

extern "C" void kernel_launch(void* const* d_in, const int* in_sizes, int n_in,
                              void* d_out, int out_size, void* d_ws, size_t ws_size,
                              hipStream_t stream) {
  (void)in_sizes; (void)n_in; (void)out_size; (void)ws_size;
  const float* x  = (const float*)d_in[0];
  const float* wq = (const float*)d_in[1];
  const float* bq = (const float*)d_in[2];
  const float* wk = (const float*)d_in[3];
  const float* bk = (const float*)d_in[4];
  const float* wv = (const float*)d_in[5];
  const float* bv = (const float*)d_in[6];
  const float* wo = (const float*)d_in[7];
  const float* bo = (const float*)d_in[8];

  const size_t NX = (size_t)Bsz * Tn * Dm;        // 6,291,456
  const size_t NW = (size_t)Dm * Dm;              // 589,824
  const size_t NHd = (size_t)Bsz * NHh * Tn * DHd; // == NX

  char* w = (char*)d_ws;
  auto takeBF = [&](size_t n) -> __bf16* {
    __bf16* p = (__bf16*)w;
    w += ((n * 2 + 255) & ~(size_t)255);
    return p;
  };
  __bf16* xb   = takeBF(NX);
  __bf16* wqb  = takeBF(NW);
  __bf16* wkb  = takeBF(NW);
  __bf16* wvb  = takeBF(NW);
  __bf16* wob  = takeBF(NW);
  __bf16* qb   = takeBF(NHd);
  __bf16* kbuf = takeBF(NHd);
  __bf16* vtb  = takeBF(NHd);
  __bf16* zbuf = takeBF(NX);

  const int CT = 256;
  cvt_f32_bf16<<<(int)((NX + CT - 1) / CT), CT, 0, stream>>>(x,  xb,  (int)NX);
  cvt_f32_bf16<<<(int)((NW + CT - 1) / CT), CT, 0, stream>>>(wq, wqb, (int)NW);
  cvt_f32_bf16<<<(int)((NW + CT - 1) / CT), CT, 0, stream>>>(wk, wkb, (int)NW);
  cvt_f32_bf16<<<(int)((NW + CT - 1) / CT), CT, 0, stream>>>(wv, wvb, (int)NW);
  cvt_f32_bf16<<<(int)((NW + CT - 1) / CT), CT, 0, stream>>>(wo, wob, (int)NW);

  qkv_gemm_wmma<<<dim3(64, 12, 3), 128, 0, stream>>>(xb, wqb, wkb, wvb, bq, bk, bv,
                                                     qb, kbuf, vtb);
  flash_attn_wmma<<<dim3(Tn / 64, Bsz * NHh), 128, 0, stream>>>(qb, kbuf, vtb, zbuf);
  oproj_gemm_wmma<<<dim3(64, 12), 128, 0, stream>>>(zbuf, wob, bo, (float*)d_out);
}
